// CustomMultiLossLayer_4707284157105
// MI455X (gfx1250) — compile-verified, matrix operand order checked by reading, not run on hardware
//
#include <hip/hip_runtime.h>

// ---------------------------------------------------------------------------
// CDNA5 / gfx1250 implementation of CustomMultiLossLayer reference.
//
// Pipeline (all on `stream`, graph-capturable, deterministic):
//   1. init_keys     : descending-float radix keys + identity payload
//   2. 8x { rs_hist ; ex_scan_u32 ; rs_scatter }  : stable LSD 4-bit radix sort
//   3. gather_prep   : xbeta/event gathers, exp() streams for both scans
//   4. wmma_scan256  : matmul-based inclusive scan. Preferred: exact f32 via
//                      V_WMMA_F32_16X16X4_F32 (12 wmma/tile). Fallback: f16
//                      hi/lo split via V_WMMA_F32_16X16X32_F16 (6 wmma/tile).
//                      Tile staging uses GLOBAL_LOAD_ASYNC_TO_LDS_B32 +
//                      s_wait_asynccnt (CDNA5 async engine).
//   5. ex_scan_f32   : cross-tile offsets (single block)
//   6. cox_final / ord_final : fused log/denominator + deterministic reductions
//   7. combine_kernel: multi-loss combination, writes scalar d_out[0]
// ---------------------------------------------------------------------------

typedef __attribute__((ext_vector_type(16))) _Float16 v16h;
typedef __attribute__((ext_vector_type(8)))  float    v8f;
typedef __attribute__((ext_vector_type(2)))  float    v2f;

#if __has_builtin(__builtin_amdgcn_wmma_f32_16x16x4_f32)
#define HAVE_WMMA_F32K4 1
#else
#define HAVE_WMMA_F32K4 0
#endif

#define USE_ASYNC_LDS 1

#define RS_THREADS 256
#define RS_EPT     8
#define RS_ELEMS   (RS_THREADS * RS_EPT)   // 2048 elements / sort block
#define SCAN_WAVES 4                        // 4 wave32 tiles per scan block
#define COX_BLOCKS 256
#define ORD_BLOCKS 32

// ---------------- radix sort ------------------------------------------------

__global__ __launch_bounds__(256) void init_keys(const float* __restrict__ y_true0,
                                                 unsigned* __restrict__ keys,
                                                 unsigned* __restrict__ vals, int n) {
  int stride = blockDim.x * gridDim.x;
  for (int i = blockIdx.x * blockDim.x + threadIdx.x; i < n; i += stride) {
    unsigned u = __float_as_uint(y_true0[2 * i]);              // time
    unsigned ka = u ^ ((u >> 31) ? 0xFFFFFFFFu : 0x80000000u); // ascending key
    keys[i] = ~ka;                                             // descending sort
    vals[i] = (unsigned)i;
  }
}

__global__ __launch_bounds__(RS_THREADS) void rs_hist(const unsigned* __restrict__ keys,
                                                      unsigned* __restrict__ counts,
                                                      int n, int shift, int numBlocks) {
  __shared__ unsigned h[16];
  if (threadIdx.x < 16) h[threadIdx.x] = 0u;
  __syncthreads();
  int base = blockIdx.x * RS_ELEMS + threadIdx.x * RS_EPT;
  #pragma unroll
  for (int e = 0; e < RS_EPT; ++e) {
    int i = base + e;
    if (i < n) atomicAdd(&h[(keys[i] >> shift) & 15u], 1u);
  }
  __syncthreads();
  if (threadIdx.x < 16) counts[threadIdx.x * numBlocks + blockIdx.x] = h[threadIdx.x];
}

__global__ __launch_bounds__(256) void ex_scan_u32(unsigned* __restrict__ data, int len) {
  __shared__ unsigned buf[256];
  __shared__ unsigned carry;
  int tx = threadIdx.x;
  if (tx == 0) carry = 0u;
  __syncthreads();
  for (int base = 0; base < len; base += 256) {
    int i = base + tx;
    unsigned v = (i < len) ? data[i] : 0u;
    buf[tx] = v;
    __syncthreads();
    for (int o = 1; o < 256; o <<= 1) {
      unsigned t = (tx >= o) ? buf[tx - o] : 0u;
      __syncthreads();
      buf[tx] += t;
      __syncthreads();
    }
    unsigned incl = buf[tx] + carry;
    if (i < len) data[i] = incl - v;   // exclusive
    __syncthreads();
    if (tx == 255) carry = incl;
    __syncthreads();
  }
}

// Stable scatter: each thread owns RS_EPT consecutive elements; block-level
// exclusive scan over the [digit][thread] histogram yields stable local ranks.
__global__ __launch_bounds__(RS_THREADS) void rs_scatter(const unsigned* __restrict__ keysIn,
                                                         const unsigned* __restrict__ valsIn,
                                                         unsigned* __restrict__ keysOut,
                                                         unsigned* __restrict__ valsOut,
                                                         const unsigned* __restrict__ gcounts,
                                                         int n, int shift, int numBlocks) {
  __shared__ unsigned hist[16 * RS_THREADS];   // digit-major, thread-minor
  __shared__ unsigned buf[RS_THREADS];
  __shared__ unsigned digitStart[16];
  __shared__ unsigned gbase[16];
  __shared__ unsigned carry;
  const int tx = threadIdx.x;
  const int base = blockIdx.x * RS_ELEMS + tx * RS_EPT;

  unsigned lk[RS_EPT], lv[RS_EPT], lh[16];
  #pragma unroll
  for (int d = 0; d < 16; ++d) lh[d] = 0u;
  #pragma unroll
  for (int e = 0; e < RS_EPT; ++e) {
    int i = base + e;
    if (i < n) {
      lk[e] = keysIn[i];
      lv[e] = valsIn[i];
      lh[(lk[e] >> shift) & 15u]++;
    } else {
      lk[e] = 0u; lv[e] = 0u;
    }
  }
  #pragma unroll
  for (int d = 0; d < 16; ++d) hist[d * RS_THREADS + tx] = lh[d];
  if (tx < 16) gbase[tx] = gcounts[tx * numBlocks + blockIdx.x];
  if (tx == 0) carry = 0u;
  __syncthreads();

  // exclusive scan over 16*256 entries (chunk c == digit c)
  for (int c = 0; c < 16; ++c) {
    unsigned v = hist[c * RS_THREADS + tx];
    buf[tx] = v;
    __syncthreads();
    for (int o = 1; o < 256; o <<= 1) {
      unsigned t = (tx >= o) ? buf[tx - o] : 0u;
      __syncthreads();
      buf[tx] += t;
      __syncthreads();
    }
    unsigned incl = buf[tx] + carry;
    hist[c * RS_THREADS + tx] = incl - v;
    __syncthreads();
    if (tx == 255) carry = incl;
    __syncthreads();
  }
  if (tx < 16) digitStart[tx] = hist[tx * RS_THREADS];  // elems with smaller digit
  __syncthreads();

  unsigned lc[16];
  #pragma unroll
  for (int d = 0; d < 16; ++d) lc[d] = 0u;
  #pragma unroll
  for (int e = 0; e < RS_EPT; ++e) {
    int i = base + e;
    if (i >= n) continue;
    unsigned d = (lk[e] >> shift) & 15u;
    unsigned rank = hist[d * RS_THREADS + tx] - digitStart[d] + lc[d];
    lc[d]++;
    unsigned pos = gbase[d] + rank;
    keysOut[pos] = lk[e];
    valsOut[pos] = lv[e];
  }
}

// ---------------- gathers ---------------------------------------------------

__global__ __launch_bounds__(256) void gather_prep(const unsigned* __restrict__ order,
                                                   const float* __restrict__ y_true0,
                                                   const float* __restrict__ y_pred0,
                                                   const float* __restrict__ y_pred1,
                                                   const int* __restrict__ Hj,
                                                   float* __restrict__ xb0s,
                                                   float* __restrict__ ev,
                                                   float* __restrict__ e0,
                                                   float* __restrict__ aArr,
                                                   float* __restrict__ bArr,
                                                   int n, int m) {
  int stride = blockDim.x * gridDim.x;
  for (int i = blockIdx.x * blockDim.x + threadIdx.x; i < n; i += stride) {
    unsigned o = order[i];
    float x = y_pred0[o];
    xb0s[i] = x;
    ev[i]   = y_true0[2 * o + 1];
    e0[i]   = expf(x);
  }
  for (int j = blockIdx.x * blockDim.x + threadIdx.x; j < m; j += stride) {
    unsigned o = order[Hj[j]];
    float x = y_pred1[o];
    aArr[j] = expf(x);
    bArr[j] = expf(-x);
  }
}

// ---------------- WMMA matmul-scan ------------------------------------------
// f16 A-matrix 16x32 VGPR map (ISA 7.12.2): M = lane%16,
//   half h -> vgpr v=h/2, j=h%2, hi=lane/16:
//   K = (v<4 ? 2v+j : 16+2(v-4)+j) + 8*hi
__device__ __forceinline__ int wmma_k_of(int h, int lane) {
  int v = h >> 1, j = h & 1, hi = lane >> 4;
  int k = (v < 4) ? (2 * v + j) : (16 + 2 * (v - 4) + j);
  return k + 8 * hi;
}

// One wave32 computes the inclusive scan of a 256-element tile via
//   scan = X*U + Lstrict*(X*J)    (X row-major 16x16, U/L/J 0-1 matrices)
__global__ __launch_bounds__(32 * SCAN_WAVES) void wmma_scan256(const float* __restrict__ in,
                                                                float* __restrict__ out,
                                                                float* __restrict__ tileSums,
                                                                int n, int numTiles) {
  __shared__ float xs[SCAN_WAVES][256];
  __shared__ float rs[SCAN_WAVES][16];
  const int lane = threadIdx.x & 31;
  const int wave = threadIdx.x >> 5;
  const int tile = blockIdx.x * SCAN_WAVES + wave;
  const int base = tile * 256;

#if USE_ASYNC_LDS && defined(__gfx1250__)
  // Stage the tile into LDS through the CDNA5 async path (no VGPR round-trip).
  // Out-of-range lanes clamp the *global address* instead of zero-filling:
  // padding garbage is benign (see analysis: scan@p only reads elems <= p,
  // outputs >= n are never stored, last-tile sum feeds no valid tile).
  #pragma unroll
  for (int k = 0; k < 8; ++k) {
    int i = base + lane + 32 * k;
    int ci = (i < n) ? i : (n - 1);
    if (ci < 0) ci = 0;
    unsigned ldsOff = (unsigned)(size_t)&xs[wave][lane + 32 * k];
    unsigned long long ga = (unsigned long long)(size_t)(in + ci);
    asm volatile("global_load_async_to_lds_b32 %0, %1, off"
                 :: "v"(ldsOff), "v"(ga) : "memory");
  }
  asm volatile("s_wait_asynccnt 0x0" ::: "memory");
#else
  #pragma unroll
  for (int k = 0; k < 8; ++k) {
    int i = base + lane + 32 * k;
    float v = 0.f;
    if (tile < numTiles && i < n) v = in[i];
    xs[wave][lane + 32 * k] = v;
  }
#endif
  __syncthreads();

  const int mn = lane & 15;   // A's M and B's N
  const int hi = lane >> 4;

#if HAVE_WMMA_F32K4
  // ---- exact f32 path: V_WMMA_F32_16X16X4_F32, K split into 4 chunks ----
  // f32 A 16x4 map: M=lane%16, K = vgpr + 2*(lane/16); B assumed mirror.
  v8f d1 = {};   // X*U : within-row inclusive prefix
  v8f d2 = {};   // X*J : row sums broadcast
  #pragma unroll
  for (int kb = 0; kb < 4; ++kb) {
    v2f a, bu, bj;
    #pragma unroll
    for (int v = 0; v < 2; ++v) {
      int K = kb * 4 + v + 2 * hi;
      a[v]  = xs[wave][mn * 16 + K];
      bu[v] = (K <= mn) ? 1.f : 0.f;   // U[k][n] = 1 if k<=n
      bj[v] = 1.f;                      // all-ones
    }
    d1 = __builtin_amdgcn_wmma_f32_16x16x4_f32(false, a, false, bu, (short)0, d1, false, false);
    d2 = __builtin_amdgcn_wmma_f32_16x16x4_f32(false, a, false, bj, (short)0, d2, false, false);
  }
  // publish row sums from column N==0 holders (C/D: M = vgpr + 8*(lane/16))
  if ((lane & 15) == 0) {
    #pragma unroll
    for (int v = 0; v < 8; ++v) rs[wave][v + 8 * hi] = d2[v];
  }
  __syncthreads();
  v8f d3 = d1;   // += Lstrict*S  -> full inclusive scan of the tile
  #pragma unroll
  for (int kb = 0; kb < 4; ++kb) {
    v2f al, bs;
    #pragma unroll
    for (int v = 0; v < 2; ++v) {
      int K = kb * 4 + v + 2 * hi;
      al[v] = (K < mn) ? 1.f : 0.f;    // strict lower [m][k]
      bs[v] = rs[wave][K];
    }
    d3 = __builtin_amdgcn_wmma_f32_16x16x4_f32(false, al, false, bs, (short)0, d3, false, false);
  }
#else
  // ---- fallback: f16 inputs split hi/lo for near-f32 accuracy ----
  v16h aXhi, aXlo, bU, bJ, aL;
  #pragma unroll
  for (int h = 0; h < 16; ++h) {
    int K = wmma_k_of(h, lane);
    float xv = (K < 16) ? xs[wave][mn * 16 + K] : 0.f;
    _Float16 hif = (_Float16)xv;
    aXhi[h] = hif;
    aXlo[h] = (_Float16)(xv - (float)hif);
    bU[h] = (_Float16)((K <= mn) ? 1.f : 0.f);
    bJ[h] = (_Float16)((K < 16) ? 1.f : 0.f);
    aL[h] = (_Float16)((K < mn) ? 1.f : 0.f);
  }
  v8f zero = {};
  v8f d1 = __builtin_amdgcn_wmma_f32_16x16x32_f16(false, aXhi, false, bU, (short)0, zero, false, false);
  d1     = __builtin_amdgcn_wmma_f32_16x16x32_f16(false, aXlo, false, bU, (short)0, d1,   false, false);
  v8f d2 = __builtin_amdgcn_wmma_f32_16x16x32_f16(false, aXhi, false, bJ, (short)0, zero, false, false);
  d2     = __builtin_amdgcn_wmma_f32_16x16x32_f16(false, aXlo, false, bJ, (short)0, d2,   false, false);
  if ((lane & 15) == 0) {
    #pragma unroll
    for (int v = 0; v < 8; ++v) rs[wave][v + 8 * hi] = d2[v];
  }
  __syncthreads();
  v16h bShi, bSlo;
  #pragma unroll
  for (int h = 0; h < 16; ++h) {
    int K = wmma_k_of(h, lane);
    float sv = (K < 16) ? rs[wave][K] : 0.f;
    _Float16 hif = (_Float16)sv;
    bShi[h] = hif;
    bSlo[h] = (_Float16)(sv - (float)hif);
  }
  v8f d3 = __builtin_amdgcn_wmma_f32_16x16x32_f16(false, aL, false, bShi, (short)0, d1, false, false);
  d3     = __builtin_amdgcn_wmma_f32_16x16x32_f16(false, aL, false, bSlo, (short)0, d3, false, false);
#endif

  if (tile < numTiles) {
    #pragma unroll
    for (int v = 0; v < 8; ++v) {
      int m = v + 8 * hi;
      int i = base + m * 16 + (lane & 15);
      if (i < n) out[i] = d3[v];
    }
    if (lane == 0) {
      float t = 0.f;
      #pragma unroll
      for (int k = 0; k < 16; ++k) t += rs[wave][k];
      tileSums[tile] = t;
    }
  }
}

__global__ __launch_bounds__(256) void ex_scan_f32(float* __restrict__ data, int len) {
  __shared__ float buf[256];
  __shared__ float carry;
  int tx = threadIdx.x;
  if (tx == 0) carry = 0.f;
  __syncthreads();
  for (int base = 0; base < len; base += 256) {
    int i = base + tx;
    float v = (i < len) ? data[i] : 0.f;
    buf[tx] = v;
    __syncthreads();
    for (int o = 1; o < 256; o <<= 1) {
      float t = (tx >= o) ? buf[tx - o] : 0.f;
      __syncthreads();
      buf[tx] += t;
      __syncthreads();
    }
    float incl = buf[tx] + carry;
    if (i < len) data[i] = incl - v;
    __syncthreads();
    if (tx == 255) carry = incl;
    __syncthreads();
  }
}

// ---------------- finalize --------------------------------------------------

__global__ __launch_bounds__(256) void cox_final(const float* __restrict__ xb,
                                                 const float* __restrict__ ev,
                                                 const float* __restrict__ scanN,
                                                 const float* __restrict__ tileOff,
                                                 int n, float* __restrict__ partial) {
  float acc = 0.f;
  int stride = blockDim.x * gridDim.x;
  for (int i = blockIdx.x * blockDim.x + threadIdx.x; i < n; i += stride) {
    float denom = scanN[i] + tileOff[i >> 8];
    acc -= ev[i] * (xb[i] - logf(denom));
  }
  __shared__ float s[256];
  s[threadIdx.x] = acc;
  __syncthreads();
  for (int o = 128; o > 0; o >>= 1) {
    if (threadIdx.x < o) s[threadIdx.x] += s[threadIdx.x + o];
    __syncthreads();
  }
  if (threadIdx.x == 0) partial[blockIdx.x] = s[0];
}

__global__ __launch_bounds__(256) void ord_final(const float* __restrict__ bArr,
                                                 const float* __restrict__ scanM,
                                                 const float* __restrict__ tileOff,
                                                 int m, float* __restrict__ partial) {
  float acc = 0.f;
  int stride = blockDim.x * gridDim.x;
  for (int j = blockIdx.x * blockDim.x + threadIdx.x; j < m; j += stride) {
    acc += bArr[j] * (scanM[j] + tileOff[j >> 8]);
  }
  __shared__ float s[256];
  s[threadIdx.x] = acc;
  __syncthreads();
  for (int o = 128; o > 0; o >>= 1) {
    if (threadIdx.x < o) s[threadIdx.x] += s[threadIdx.x + o];
    __syncthreads();
  }
  if (threadIdx.x == 0) partial[blockIdx.x] = s[0];
}

__global__ __launch_bounds__(256) void combine_kernel(const float* __restrict__ coxPartial,
                                                      const float* __restrict__ ordPartial,
                                                      const float* __restrict__ log_vars,
                                                      int n, int m, float* __restrict__ out) {
  __shared__ float s[256];
  int tx = threadIdx.x;
  s[tx] = coxPartial[tx];           // COX_BLOCKS == 256
  __syncthreads();
  for (int o = 128; o > 0; o >>= 1) {
    if (tx < o) s[tx] += s[tx + o];
    __syncthreads();
  }
  float sumCox = s[0];
  __syncthreads();
  s[tx] = (tx < ORD_BLOCKS) ? ordPartial[tx] : 0.f;
  __syncthreads();
  for (int o = 128; o > 0; o >>= 1) {
    if (tx < o) s[tx] += s[tx + o];
    __syncthreads();
  }
  if (tx == 0) {
    float sumOrd = s[0];
    float lv0 = log_vars[0];
    float lv1 = log_vars[1];
    float prec1 = fminf(fmaxf(expf(-lv1), 0.f), 1.f);
    float pairs = (float)(0.5 * (double)m * (double)(m + 1));
    float cost2 = pairs - sumOrd;   // sum over upper-tri of (1 - exp(dx))
    out[0] = sumCox + (float)n * lv0 + prec1 * cost2 + lv1;
  }
}

// ---------------- host side -------------------------------------------------

extern "C" void kernel_launch(void* const* d_in, const int* in_sizes, int n_in,
                              void* d_out, int out_size, void* d_ws, size_t ws_size,
                              hipStream_t stream) {
  (void)n_in; (void)out_size; (void)ws_size;
  const float* y_true0  = (const float*)d_in[0];
  const float* y_pred0  = (const float*)d_in[2];
  const float* y_pred1  = (const float*)d_in[3];
  const int*   Hj       = (const int*)d_in[4];
  const float* log_vars = (const float*)d_in[5];
  const int N = in_sizes[0] / 2;
  const int M = in_sizes[4];

  char* ws = (char*)d_ws;
  size_t off = 0;
  auto carve = [&](size_t bytes) -> void* {
    void* p = ws + off;
    off += (bytes + 255) & ~(size_t)255;
    return p;
  };

  const int nb = (N + RS_ELEMS - 1) / RS_ELEMS;
  unsigned* keysA  = (unsigned*)carve((size_t)N * 4);
  unsigned* valsA  = (unsigned*)carve((size_t)N * 4);
  unsigned* keysB  = (unsigned*)carve((size_t)N * 4);
  unsigned* valsB  = (unsigned*)carve((size_t)N * 4);
  unsigned* counts = (unsigned*)carve((size_t)16 * nb * 4);
  float* xb0s  = (float*)carve((size_t)N * 4);
  float* ev    = (float*)carve((size_t)N * 4);
  float* e0    = (float*)carve((size_t)N * 4);
  float* scanN = (float*)carve((size_t)N * 4);
  float* aArr  = (float*)carve((size_t)M * 4);
  float* bArr  = (float*)carve((size_t)M * 4);
  float* scanM = (float*)carve((size_t)M * 4);
  const int tilesN = (N + 255) / 256;
  const int tilesM = (M + 255) / 256;
  float* tsum0 = (float*)carve((size_t)tilesN * 4);
  float* tsum1 = (float*)carve((size_t)tilesM * 4);
  float* coxPartial = (float*)carve((size_t)COX_BLOCKS * 4);
  float* ordPartial = (float*)carve((size_t)ORD_BLOCKS * 4);

  // 1) keys
  init_keys<<<256, 256, 0, stream>>>(y_true0, keysA, valsA, N);

  // 2) stable LSD radix sort, 8 x 4-bit, ends back in keysA/valsA
  unsigned *ki = keysA, *vi = valsA, *ko = keysB, *vo = valsB;
  for (int p = 0; p < 8; ++p) {
    int shift = 4 * p;
    rs_hist<<<nb, RS_THREADS, 0, stream>>>(ki, counts, N, shift, nb);
    ex_scan_u32<<<1, 256, 0, stream>>>(counts, 16 * nb);
    rs_scatter<<<nb, RS_THREADS, 0, stream>>>(ki, vi, ko, vo, counts, N, shift, nb);
    unsigned* tk = ki; ki = ko; ko = tk;
    unsigned* tv = vi; vi = vo; vo = tv;
  }
  const unsigned* order = vi;   // = valsA after 8 passes

  // 3) gathers + exp streams
  gather_prep<<<256, 256, 0, stream>>>(order, y_true0, y_pred0, y_pred1, Hj,
                                       xb0s, ev, e0, aArr, bArr, N, M);

  // 4) WMMA matmul-scans (per-256 tiles)
  wmma_scan256<<<(tilesN + SCAN_WAVES - 1) / SCAN_WAVES, 32 * SCAN_WAVES, 0, stream>>>(
      e0, scanN, tsum0, N, tilesN);
  wmma_scan256<<<(tilesM + SCAN_WAVES - 1) / SCAN_WAVES, 32 * SCAN_WAVES, 0, stream>>>(
      aArr, scanM, tsum1, M, tilesM);

  // 5) cross-tile exclusive offsets
  ex_scan_f32<<<1, 256, 0, stream>>>(tsum0, tilesN);
  ex_scan_f32<<<1, 256, 0, stream>>>(tsum1, tilesM);

  // 6) deterministic two-stage reductions
  cox_final<<<COX_BLOCKS, 256, 0, stream>>>(xb0s, ev, scanN, tsum0, N, coxPartial);
  ord_final<<<ORD_BLOCKS, 256, 0, stream>>>(bArr, scanM, tsum1, M, ordPartial);

  // 7) combine
  combine_kernel<<<1, 256, 0, stream>>>(coxPartial, ordPartial, log_vars, N, M,
                                        (float*)d_out);
}
